// MiniRocketFeatures_58652073394407
// MI455X (gfx1250) — matrix operand order repaired and checked by reading, not compiled, not run on hardware
//
#include <hip/hip_runtime.h>
#include <hip/hip_bf16.h>

typedef __attribute__((ext_vector_type(16))) _Float16 v16h;
typedef __attribute__((ext_vector_type(8)))  _Float16 v8h;
typedef __attribute__((ext_vector_type(4)))  _Float16 v4h;
typedef __attribute__((ext_vector_type(8)))  float    v8f;

#define NUM_KERNELS 84
#define C_IN        9
#define KSIZE       9
#define SEQ_LEN     2048
#define BATCH       64
#define NUM_DIL     26
#define MAXNF       15
#define NFPAD       16
#define KDIM        81
#define KPAD        96
#define MPAD        96
#define NTILE       128
#define VSTRIDE     104     // halves; 208B row pitch, 16B aligned
#define FEAT_TOTAL  9996
#define NTHREADS    384     // 12 waves: 6 mtiles x 2 column-halves

// Static MiniRocket config for SEQ_LEN=2048 (mirrors _set_dilations exactly).
__constant__ int c_dil[NUM_DIL]  = {1,2,3,4,5,7,8,10,12,14,17,20,25,29,35,42,
                                    51,61,73,87,104,125,149,179,214,255};
__constant__ int c_base[NUM_DIL] = {0,1260,2268,2604,2940,3612,3948,4284,4620,
                                    4956,5292,5628,5964,6300,6636,6972,7308,
                                    7644,7980,8232,8484,8736,8988,9240,9492,9744};

// A 16x32 f16 frag: lane holds row M=lane%16; lanes 0-15 K {0..7,16..23}+kc*32,
// lanes 16-31 K {8..15,24..31}+kc*32  (ISA 7.12.2)
__device__ inline v16h load_a_frag(const _Float16* A, int mtile, int kc, int lane) {
    int m  = mtile * 16 + (lane & 15);
    int hi = (lane >> 4) & 1;
    const _Float16* p = A + m * KPAD + kc * 32 + hi * 8;
    v8h s0 = *(const v8h*)(p);
    v8h s1 = *(const v8h*)(p + 16);
    v16h r;
#pragma unroll
    for (int i = 0; i < 8; ++i) { r[i] = s0[i]; r[i + 8] = s1[i]; }
    return r;
}

// B 32x16 f16 frag: lane holds col N=lane%16; lanes 0-15 K {0..7,16..23}, 16-31 K {8..15,24..31}.
__device__ inline v16h load_b_frag(const _Float16* V, int cts, int kc, int lane) {
    int n = lane & 15;
    const _Float16* p = V + (cts * 16 + n) * VSTRIDE + kc * 32 + ((lane >> 4) << 3);
    v8h s0 = *(const v8h*)(p);
    v8h s1 = *(const v8h*)(p + 16);
    v16h r;
#pragma unroll
    for (int i = 0; i < 8; ++i) { r[i] = s0[i]; r[i + 8] = s1[i]; }
    return r;
}

template <int NF>
__global__ __launch_bounds__(NTHREADS)
void mr_kernel(const float* __restrict__ x,
               const float* __restrict__ kern,
               const float* __restrict__ ccomb,
               const float* __restrict__ biases,
               float* __restrict__ out) {
    constexpr int NG = (NF + 3) / 4;          // packed counter groups (4 bytes each)
    constexpr bool VB = (NF <= 8);            // bias in VGPRs vs LDS float4 loads

    __shared__ alignas(16) _Float16 lds_A[MPAD * KPAD];       // 18 KB
    __shared__ alignas(16) _Float16 lds_V[NTILE * VSTRIDE];   // 26 KB
    __shared__ unsigned lds_cnt[MPAD * NFPAD];                // 6 KB
    __shared__ alignas(16) float lds_bias[VB ? 16 : MPAD * NFPAD];

    // dilation index from template group + blockIdx.x
    int d;
    if constexpr (NF == 15)      d = 0;
    else if constexpr (NF == 12) d = 1;
    else if constexpr (NF == 8)  d = 4;
    else if constexpr (NF == 4)  d = (blockIdx.x < 2) ? (int)blockIdx.x + 2
                                                      : (int)blockIdx.x + 3;
    else                         d = 18 + (int)blockIdx.x;

    const int b    = blockIdx.y;
    const int tid  = threadIdx.x;
    const int lane = tid & 31;
    const int wave = tid >> 5;                // 0..11
    const int mtile = wave >> 1;              // 0..5
    const int half  = wave & 1;               // column half
    const int hi    = (lane >> 4) & 1;
    const int n     = lane & 15;
    const int dil  = c_dil[d];
    const int pad  = 4 * dil;
    const int pad1 = d & 1;

    // ---- stage A = mask[d,c,k]*kern[k,j] (f16, zero padded to 96x96) ----
    for (int idx = tid; idx < MPAD * KPAD; idx += NTHREADS) {
        int m = idx / KPAD, kk = idx - m * KPAD;
        float v = 0.0f;
        if (m < NUM_KERNELS && kk < KDIM) {
            int c = kk / KSIZE, j = kk - c * KSIZE;
            v = ccomb[(d * C_IN + c) * NUM_KERNELS + m] * kern[m * KSIZE + j];
        }
        lds_A[idx] = (_Float16)v;
    }
    if constexpr (!VB) {  // LDS bias table, +INF on invalid rows
        for (int idx = tid; idx < MPAD * NFPAD; idx += NTHREADS) {
            int k = idx >> 4, f = idx & 15;
            lds_bias[idx] = (k < NUM_KERNELS && f < MAXNF)
                          ? biases[(d * NUM_KERNELS + k) * MAXNF + f] : 3.0e38f;
        }
    }
    for (int idx = tid; idx < MPAD * NFPAD; idx += NTHREADS) lds_cnt[idx] = 0u;
    __syncthreads();

    // ---- per-wave invariants: A fragments + (small-NF) bias registers ----
    v16h a0 = load_a_frag(lds_A, mtile, 0, lane);
    v16h a1 = load_a_frag(lds_A, mtile, 1, lane);
    v16h a2 = load_a_frag(lds_A, mtile, 2, lane);

    float bv[8][VB ? NF : 1];
    if constexpr (VB) {
#pragma unroll
        for (int r = 0; r < 8; ++r) {
            int kL = mtile * 16 + r + 8 * hi;
#pragma unroll
            for (int f = 0; f < NF; ++f)
                bv[r][f] = (kL < NUM_KERNELS)
                         ? biases[(d * NUM_KERNELS + kL) * MAXNF + f] : 3.0e38f;
        }
    }

    unsigned cnt[8][NG];
#pragma unroll
    for (int r = 0; r < 8; ++r)
#pragma unroll
        for (int g = 0; g < NG; ++g) cnt[r][g] = 0u;

    const float* xb = x + (size_t)b * C_IN * SEQ_LEN;

    for (int it = 0; it < SEQ_LEN / NTILE; ++it) {
        const int tbase = it * NTILE;
        __syncthreads();
        // ---- stage V: 4 K-halves per ds_store_b64, coalesced t reads ----
        for (int idx = tid; idx < (KPAD / 4) * NTILE; idx += NTHREADS) {
            int col = idx & (NTILE - 1), grp = idx >> 7;
            int kk0 = grp * 4;
            v4h v;
#pragma unroll
            for (int e = 0; e < 4; ++e) {
                int kk = kk0 + e;
                float fv = 0.0f;
                if (kk < KDIM) {
                    int c = kk / KSIZE, j = kk - c * KSIZE;
                    int t = tbase + col + (j - 4) * dil;
                    if (t >= 0 && t < SEQ_LEN) fv = xb[c * SEQ_LEN + t];
                }
                v[e] = (_Float16)fv;
            }
            *(v4h*)(&lds_V[col * VSTRIDE + kk0]) = v;
        }
        __syncthreads();

#pragma unroll
        for (int ct = 0; ct < 4; ++ct) {
            const int cts = half * 4 + ct;               // col tile in slab
            const int t   = tbase + cts * 16 + n;        // this lane's column
            const bool in_inner = (t >= pad) && (t < SEQ_LEN - pad);
            const bool vEven = (pad1 == 0) || in_inner;  // rows with even k
            const bool vOdd  = (pad1 == 1) || in_inner;  // rows with odd k

            v16h b0 = load_b_frag(lds_V, cts, 0, lane);
            v16h b1 = load_b_frag(lds_V, cts, 1, lane);
            v16h b2 = load_b_frag(lds_V, cts, 2, lane);
            v8f acc = {};
            acc = __builtin_amdgcn_wmma_f32_16x16x32_f16(false, a0, false, b0,
                                                         (short)0, acc, false, false);
            acc = __builtin_amdgcn_wmma_f32_16x16x32_f16(false, a1, false, b1,
                                                         (short)0, acc, false, false);
            acc = __builtin_amdgcn_wmma_f32_16x16x32_f16(false, a2, false, b2,
                                                         (short)0, acc, false, false);

            // ---- fused PPV: pure per-lane VALU, byte-packed counters ----
#pragma unroll
            for (int r = 0; r < 8; ++r) {
                const bool ok = ((r & 1) == 0) ? vEven : vOdd;
                const float cvm = ok ? acc[r] : -3.0e38f;
                if constexpr (VB) {
#pragma unroll
                    for (int f = 0; f < NF; ++f)
                        cnt[r][f >> 2] += ((cvm > bv[r][f]) ? 1u : 0u) << ((f & 3) * 8);
                } else {
                    const int kL = mtile * 16 + r + 8 * hi;
                    const float4* bp = (const float4*)&lds_bias[kL * NFPAD];
#pragma unroll
                    for (int g = 0; g < NG; ++g) {
                        float4 bb = bp[g];
                        cnt[r][g] += ((cvm > bb.x) ? 1u : 0u)
                                   + (((cvm > bb.y) ? 1u : 0u) << 8)
                                   + (((cvm > bb.z) ? 1u : 0u) << 16)
                                   + (((cvm > bb.w) ? 1u : 0u) << 24);
                    }
                }
            }
        }
    }

    // ---- epilogue: cross-lane reduce (u16 repack) + few LDS atomics ----
#pragma unroll
    for (int r = 0; r < 8; ++r) {
        const int kRow = mtile * 16 + r + 8 * hi;
#pragma unroll
        for (int g = 0; g < NG; ++g) {
            unsigned c  = cnt[r][g];
            unsigned lo = c & 0x00FF00FFu;           // f%4==0 (lo16), f%4==2 (hi16)
            unsigned hh = (c >> 8) & 0x00FF00FFu;    // f%4==1, f%4==3
            lo += __shfl_xor(lo, 1); lo += __shfl_xor(lo, 2);
            lo += __shfl_xor(lo, 4); lo += __shfl_xor(lo, 8);
            hh += __shfl_xor(hh, 1); hh += __shfl_xor(hh, 2);
            hh += __shfl_xor(hh, 4); hh += __shfl_xor(hh, 8);
            if (n == 0) {
                const int f0 = g * 4;
                atomicAdd(&lds_cnt[kRow * NFPAD + f0 + 0], lo & 0xFFFFu);
                atomicAdd(&lds_cnt[kRow * NFPAD + f0 + 1], hh & 0xFFFFu);
                atomicAdd(&lds_cnt[kRow * NFPAD + f0 + 2], lo >> 16);
                atomicAdd(&lds_cnt[kRow * NFPAD + f0 + 3], hh >> 16);
            }
        }
    }
    __syncthreads();

    // ---- normalize + scatter to reference feature layout ----
    const int inner_len = SEQ_LEN - 2 * pad;
    const int base = c_base[d];
    for (int idx = tid; idx < NUM_KERNELS * NFPAD; idx += NTHREADS) {
        int k = idx >> 4, f = idx & 15;
        if (f >= NF) continue;
        const bool use_inner = ((k & 1) != pad1);
        const float norm = use_inner ? (float)inner_len : (float)SEQ_LEN;
        const float val = (float)lds_cnt[k * NFPAD + f] / norm;
        int pos;
        if ((k & 1) == pad1) pos = ((k - pad1) >> 1) * NF + f;                // block 0
        else                 pos = 42 * NF + ((k - (1 - pad1)) >> 1) * NF + f;// block 1
        out[(size_t)b * FEAT_TOTAL + base + pos] = val;
    }
}

extern "C" void kernel_launch(void* const* d_in, const int* in_sizes, int n_in,
                              void* d_out, int out_size, void* d_ws, size_t ws_size,
                              hipStream_t stream) {
    (void)in_sizes; (void)n_in; (void)d_ws; (void)ws_size; (void)out_size;
    const float* x       = (const float*)d_in[0];   // [64,9,2048]
    const float* kern    = (const float*)d_in[1];   // [756,1,9] (first 84 rows used)
    const float* ccomb   = (const float*)d_in[2];   // [26,9,84]
    const float* biases  = (const float*)d_in[3];   // [26,84,15]
    float* out           = (float*)d_out;           // [64,9996]

    dim3 block(NTHREADS);
    mr_kernel<15><<<dim3(1,  BATCH), block, 0, stream>>>(x, kern, ccomb, biases, out);
    mr_kernel<12><<<dim3(1,  BATCH), block, 0, stream>>>(x, kern, ccomb, biases, out);
    mr_kernel<8 ><<<dim3(1,  BATCH), block, 0, stream>>>(x, kern, ccomb, biases, out);
    mr_kernel<4 ><<<dim3(15, BATCH), block, 0, stream>>>(x, kern, ccomb, biases, out);
    mr_kernel<3 ><<<dim3(8,  BATCH), block, 0, stream>>>(x, kern, ccomb, biases, out);
}
